// GIN_44659069944137
// MI455X (gfx1250) — matrix operand order repaired and checked by reading, not compiled, not run on hardware
//
#include <hip/hip_runtime.h>

#define N_NODES 50000
#define N_EDGES 800000
#define D 128
#define N_LAYERS 5
#define BN_EPS 1e-5f

#define M_TILE 80                      // 50000 = 625 * 80 exactly -> no bounds checks
#define BLOCK_GEMM 320                 // 10 waves (wave32)
#define N_BLOCKS_GEMM (N_NODES / M_TILE)

typedef __attribute__((ext_vector_type(16))) _Float16 v16h;
typedef __attribute__((ext_vector_type(8)))  _Float16 v8h;
typedef __attribute__((ext_vector_type(8)))  float    v8f;

// Shared-memory plan (54,272 B static; Hs aliases Wf/Xs after a barrier):
//   [0      , 32768)  Wf  : W in WMMA-B fragment-major f16   (32 KB)
//   [32768  , 53248)  Xs  : X tile row-major f16             (20 KB)
//   [53248  , 54272)  red : BN partial column sums           ( 1 KB)
//   [0      , 40960)  Hs  : f32 output tile (reused region)  (40 KB)
#define SMEM_W  0
#define SMEM_X  32768
#define SMEM_R  (32768 + 20480)
#define SMEM_SZ (32768 + 20480 + 1024)

// ---------------------------------------------------------------- agg = x copy
__global__ void copy_f4(float4* __restrict__ dst, const float4* __restrict__ src, int n4) {
    int i = blockIdx.x * blockDim.x + threadIdx.x;
    if (i < n4) dst[i] = src[i];
}

// ------------------------------------------------- agg[dst] += x[src] (atomics)
// 32 threads (1 wave) per edge, 4 floats/thread: float4 gather + 4x global_atomic_add_f32.
// Working set is L2-resident (77 MB << 192 MB), so atomics resolve in L2.
__global__ void scatter_add(float* __restrict__ agg, const float* __restrict__ x,
                            const long long* __restrict__ src,
                            const long long* __restrict__ dst) {
    int t = blockIdx.x * blockDim.x + threadIdx.x;
    int e = t >> 5;
    if (e >= N_EDGES) return;
    int lane = t & 31;
    if (lane == 0 && e + 2048 < N_EDGES) {           // stream-prefetch the index arrays
        __builtin_prefetch(src + e + 2048, 0, 0);    // -> global_prefetch_b8
        __builtin_prefetch(dst + e + 2048, 0, 0);
    }
    int q = lane << 2;
    long long s = src[e], d = dst[e];
    const float4 v = *(const float4*)(x + s * D + q);
    float* p = agg + d * D + q;
    atomicAdd(p + 0, v.x);
    atomicAdd(p + 1, v.y);
    atomicAdd(p + 2, v.z);
    atomicAdd(p + 3, v.w);
}

// ------- one-shot: all 10 W[k][n] (f32 row-major) -> fragment-major f16 -------
// Fragment layout: Wf[(frag*32 + lane)*16 + i], frag = (k>>5)*8 + (n>>4),
// lane = ((k>>4)&1)*16 + (n&15), i = k&15.
__global__ void swizzle_W(const float* __restrict__ Wa, const float* __restrict__ Wb,
                          _Float16* __restrict__ WaF, _Float16* __restrict__ WbF) {
    int idx = blockIdx.x * blockDim.x + threadIdx.x;
    if (idx >= N_LAYERS * D * D) return;
    int l = idx >> 14, e = idx & 16383;
    int k = e >> 7, n = e & 127;
    int frag = ((k >> 5) << 3) | (n >> 4);
    int lane = (((k >> 4) & 1) << 4) | (n & 15);
    int off  = (l << 14) + (frag * 32 + lane) * 16 + (k & 15);
    WaF[off] = (_Float16)Wa[idx];
    WbF[off] = (_Float16)Wb[idx];
}

// -------- GEMM prologue: pre-swizzled W (L2-resident) -> LDS, 16B copies ------
__device__ __forceinline__ void load_W_frag(_Float16* Wf, const _Float16* __restrict__ Wfg,
                                            int tid) {
    uint4*       dst = (uint4*)Wf;
    const uint4* src = (const uint4*)Wfg;
    #pragma unroll
    for (int i = tid; i < D * D / 8; i += BLOCK_GEMM)   // 2048 x 16B
        dst[i] = src[i];                                // global_load_b128 -> ds_store_b128
}

// ------------- per-wave 16x128 @ 128x64 WMMA core (4 accumulators) ------------
__device__ __forceinline__ void wmma_tile_gemm(const _Float16* Wf, const _Float16* Xs,
                                               int r, int ch, int lane, v8f acc[4]) {
    const int lm = lane & 15, lg = lane >> 4;
    const _Float16* arow = &Xs[(r * 16 + lm) * D];
    #pragma unroll
    for (int kbi = 0; kbi < 4; ++kbi) {
        // A fragment: two contiguous 16-byte runs -> 2x ds_load_b128
        union { v16h v; v8h h[2]; } ua;
        ua.h[0] = *(const v8h*)(arow + kbi * 32 + (lg << 4));
        ua.h[1] = *(const v8h*)(arow + kbi * 32 + (lg << 4) + 8);
        #pragma unroll
        for (int nt = 0; nt < 4; ++nt) {
            // B fragment: contiguous 32 bytes per lane -> 2x ds_load_b128
            v16h b = *(const v16h*)(&Wf[((kbi * 8 + ch * 4 + nt) * 32 + lane) << 4]);
            acc[nt] = __builtin_amdgcn_wmma_f32_16x16x32_f16(
                false, ua.v, false, b, (short)0, acc[nt], false, false);
        }
    }
}

// -------------------------------------------- H = Xin @ W + b, + column stats
__global__ __launch_bounds__(BLOCK_GEMM)
void gemm1_bn_stats(const float* __restrict__ Xin, const _Float16* __restrict__ Wfg,
                    const float* __restrict__ bias, float* __restrict__ Hout,
                    float* __restrict__ gsum, float* __restrict__ gsumsq) {
    __shared__ __align__(64) unsigned char smem[SMEM_SZ];
    _Float16* Wf  = (_Float16*)(smem + SMEM_W);
    _Float16* Xs  = (_Float16*)(smem + SMEM_X);
    float*    red = (float*)(smem + SMEM_R);
    float*    Hs  = (float*)smem;   // valid only after the post-GEMM barrier

    const int tid = threadIdx.x;
    const size_t rowbase = (size_t)blockIdx.x * M_TILE;

    load_W_frag(Wf, Wfg, tid);
    for (int i = tid; i < M_TILE * D; i += BLOCK_GEMM)
        Xs[i] = (_Float16)Xin[rowbase * D + i];
    if (tid < 2 * D) red[tid] = 0.f;
    __syncthreads();

    const int w = tid >> 5, lane = tid & 31;
    const int r = w >> 1, ch = w & 1;          // row-tile 0..4, col-half 0..1
    const int lm = lane & 15, lg = lane >> 4;

    v8f acc[4] = {};
    wmma_tile_gemm(Wf, Xs, r, ch, lane, acc);
    __syncthreads();                            // Wf/Xs dead -> region becomes Hs

    #pragma unroll
    for (int nt = 0; nt < 4; ++nt) {
        int n = ch * 64 + nt * 16 + lm;
        float bv = bias[n];
        float s = 0.f, s2 = 0.f;
        #pragma unroll
        for (int v = 0; v < 8; ++v) {
            float val = acc[nt][v] + bv;
            Hs[(r * 16 + v + (lg << 3)) * D + n] = val;
            s += val;
            s2 += val * val;
        }
        atomicAdd(&red[n], s);                  // ds_add_f32
        atomicAdd(&red[D + n], s2);
    }
    __syncthreads();

    // coalesced float4 tile store + one global atomic per column per block
    float4*       g4 = (float4*)(Hout + rowbase * D);
    const float4* h4 = (const float4*)Hs;
    for (int i = tid; i < M_TILE * D / 4; i += BLOCK_GEMM) g4[i] = h4[i];
    if (tid < D) {
        atomicAdd(&gsum[tid], red[tid]);
        atomicAdd(&gsumsq[tid], red[D + tid]);
    }
}

// -------------------------------------------------- fold stats into scale/shift
__global__ void bn_stats(const float* __restrict__ gsum, const float* __restrict__ gsumsq,
                         const float* __restrict__ gamma, const float* __restrict__ beta,
                         float* __restrict__ scale, float* __restrict__ shift) {
    int n = threadIdx.x;
    float mu   = gsum[n] * (1.f / (float)N_NODES);
    float var  = gsumsq[n] * (1.f / (float)N_NODES) - mu * mu;
    float rstd = rsqrtf(var + BN_EPS);
    float g    = gamma[n] * rstd;
    scale[n] = g;
    shift[n] = beta[n] - mu * g;
}

// ---------------------- Out = relu( relu(BN(H)) @ W + b ) (BN+ReLU fused in load)
__global__ __launch_bounds__(BLOCK_GEMM)
void bn_relu_gemm2(const float* __restrict__ Hin, const float* __restrict__ scale,
                   const float* __restrict__ shift, const _Float16* __restrict__ Wfg,
                   const float* __restrict__ bias, float* __restrict__ Out) {
    __shared__ __align__(64) unsigned char smem[SMEM_SZ];
    _Float16* Wf = (_Float16*)(smem + SMEM_W);
    _Float16* Xs = (_Float16*)(smem + SMEM_X);
    float*    Hs = (float*)smem;

    const int tid = threadIdx.x;
    const size_t rowbase = (size_t)blockIdx.x * M_TILE;

    load_W_frag(Wf, Wfg, tid);
    for (int i = tid; i < M_TILE * D; i += BLOCK_GEMM) {
        int k = i & (D - 1);
        float v = Hin[rowbase * D + i] * scale[k] + shift[k];
        Xs[i] = (_Float16)fmaxf(v, 0.f);
    }
    __syncthreads();

    const int w = tid >> 5, lane = tid & 31;
    const int r = w >> 1, ch = w & 1;
    const int lm = lane & 15, lg = lane >> 4;

    v8f acc[4] = {};
    wmma_tile_gemm(Wf, Xs, r, ch, lane, acc);
    __syncthreads();                            // reuse LDS as Hs

    #pragma unroll
    for (int nt = 0; nt < 4; ++nt) {
        int n = ch * 64 + nt * 16 + lm;
        float bv = bias[n];
        #pragma unroll
        for (int v = 0; v < 8; ++v)
            Hs[(r * 16 + v + (lg << 3)) * D + n] = fmaxf(acc[nt][v] + bv, 0.f);
    }
    __syncthreads();

    float4*       g4 = (float4*)(Out + rowbase * D);
    const float4* h4 = (const float4*)Hs;
    for (int i = tid; i < M_TILE * D / 4; i += BLOCK_GEMM) g4[i] = h4[i];
}

extern "C" void kernel_launch(void* const* d_in, const int* in_sizes, int n_in,
                              void* d_out, int out_size, void* d_ws, size_t ws_size,
                              hipStream_t stream) {
    const float*     x0    = (const float*)d_in[0];
    const long long* ei    = (const long long*)d_in[1];   // int64 [2, N_EDGES]
    const float*     Wa    = (const float*)d_in[2];
    const float*     ba    = (const float*)d_in[3];
    const float*     gamma = (const float*)d_in[4];
    const float*     beta  = (const float*)d_in[5];
    const float*     Wb    = (const float*)d_in[6];
    const float*     bb    = (const float*)d_in[7];
    const long long* src = ei;
    const long long* dst = ei + N_EDGES;

    float* Xbuf    = (float*)d_ws;                       // 25.6 MB
    float* AGG     = Xbuf + (size_t)N_NODES * D;         // 25.6 MB
    float* Hbuf    = AGG  + (size_t)N_NODES * D;         // 25.6 MB
    float* gsum    = Hbuf + (size_t)N_NODES * D;         // 128
    float* gsumsq  = gsum + D;                           // 128
    float* bnscale = gsumsq + D;                         // 128
    float* bnshift = bnscale + D;                        // 128
    _Float16* WaF  = (_Float16*)(bnshift + D);           // 5 x 16384 f16 (160 KB)
    _Float16* WbF  = WaF + (size_t)N_LAYERS * D * D;     // 5 x 16384 f16 (160 KB)

    const int n4 = N_NODES * D / 4;

    // one-shot weight swizzle (f32 row-major -> fragment-major f16, L2-resident)
    swizzle_W<<<(N_LAYERS * D * D + 255) / 256, 256, 0, stream>>>(Wa, Wb, WaF, WbF);

    for (int l = 0; l < N_LAYERS; ++l) {
        const float* xin  = (l == 0) ? x0 : Xbuf;
        float*       xout = (l == N_LAYERS - 1) ? (float*)d_out : Xbuf;

        // agg = x, then agg[dst] += x[src]
        copy_f4<<<(n4 + 255) / 256, 256, 0, stream>>>((float4*)AGG, (const float4*)xin, n4);
        scatter_add<<<(N_EDGES * 32) / 256, 256, 0, stream>>>(AGG, xin, src, dst);

        hipMemsetAsync(gsum, 0, 2 * D * sizeof(float), stream);
        gemm1_bn_stats<<<N_BLOCKS_GEMM, BLOCK_GEMM, 0, stream>>>(
            AGG, WaF + (size_t)l * D * D, ba + (size_t)l * D, Hbuf, gsum, gsumsq);
        bn_stats<<<1, D, 0, stream>>>(gsum, gsumsq, gamma + (size_t)l * D,
                                      beta + (size_t)l * D, bnscale, bnshift);
        bn_relu_gemm2<<<N_BLOCKS_GEMM, BLOCK_GEMM, 0, stream>>>(
            Hbuf, bnscale, bnshift, WbF + (size_t)l * D * D, bb + (size_t)l * D, xout);
    }
    (void)in_sizes; (void)n_in; (void)out_size; (void)ws_size;
}
